// EARLIEST_82033875353984
// MI455X (gfx1250) — matrix operand order repaired and checked by reading, not compiled, not run on hardware
//
#include <hip/hip_runtime.h>
#include <hip/hip_bf16.h>

#define T_STEPS 1000
#define BATCH   256
#define FDIM    128
#define HDIM    256
#define NCLS    10
#define NWG     16
#define KDIM    (FDIM + HDIM)   // 384
#define EPS_C   0.05f

typedef __attribute__((ext_vector_type(16))) __bf16 v16bf;
typedef __attribute__((ext_vector_type(8)))  __bf16 v8bf;
typedef __attribute__((ext_vector_type(8)))  float  v8f;
typedef __attribute__((ext_vector_type(4)))  float  v4f;

union V16 { v16bf v; v8bf h[2]; __bf16 e[16]; };

__device__ __forceinline__ float sigf(float x) {
    return 1.0f / (1.0f + __expf(-x));
}
__device__ __forceinline__ float tanh_fast(float x) {
    // stable: 1 - 2/(e^{2x}+1); handles +/-inf saturation correctly
    return 1.0f - 2.0f / (__expf(2.0f * x) + 1.0f);
}

// ---------------------------------------------------------------------------
// Init kernel: zero per-step barrier counters and the h double buffer.
// Re-run on every kernel_launch so graph replays are deterministic.
// ---------------------------------------------------------------------------
__global__ void earliest_init(unsigned* __restrict__ barr,
                              unsigned short* __restrict__ hbuf) {
    int i = blockIdx.x * blockDim.x + threadIdx.x;
    if (i < T_STEPS) barr[i] = 0u;
    for (int j = i; j < 2 * BATCH * HDIM; j += gridDim.x * blockDim.x)
        hbuf[j] = 0;  // bf16 zero
}

// ---------------------------------------------------------------------------
// Main persistent kernel: 16 WGs (one per WGP), 256 threads (8 waves) each.
// WG w owns h-columns [16w,16w+16) (=> 64 gate columns, weights in LDS) and
// batch rows [16w,16w+16) for controller/halting/final classifier.
// ---------------------------------------------------------------------------
__global__ __launch_bounds__(256)
void earliest_main(const float* __restrict__ X,   const float* __restrict__ U,
                   const float* __restrict__ Wih, const float* __restrict__ Whh,
                   const float* __restrict__ bih, const float* __restrict__ bhh,
                   const float* __restrict__ Wc,  const float* __restrict__ bc,
                   const float* __restrict__ Wd,  const float* __restrict__ bd,
                   float* __restrict__ out,
                   unsigned short* __restrict__ hbuf_raw,  // [2][B][H] bf16
                   float* __restrict__ Shalt,              // [B][H] f32
                   unsigned* __restrict__ barr)            // [T] counters
{
    __shared__ __align__(32) __bf16 sW[64 * KDIM];   // 48KB weight slice, row-major [col][k]
    __shared__ float sWc[HDIM + 2];                  // W_c[0..255], W_c[256]=time, [257]=b_c
    __shared__ float sRed[16][17];
    __shared__ int   sHalted[16];
    __shared__ int   sCopy[16];
    __shared__ int   sBreak;

    __bf16* hbuf = reinterpret_cast<__bf16*>(hbuf_raw);

    const int wg   = blockIdx.x;          // 0..15
    const int tid  = threadIdx.x;
    const int lane = tid & 31;
    const int wv   = tid >> 5;            // wave 0..7
    const int half = lane >> 4;
    const int ln16 = lane & 15;

    // ---- one-time init: load bf16 weight slice into LDS --------------------
    for (int idx = tid; idx < 64 * KDIM; idx += 256) {
        int c = idx / KDIM, k = idx - c * KDIM;
        int gate = c >> 4;                               // 0..3 = i,f,g,o
        int gcol = gate * HDIM + wg * 16 + (c & 15);     // row of [W_ih|W_hh]
        float w = (k < FDIM) ? Wih[(size_t)gcol * FDIM + k]
                             : Whh[(size_t)gcol * HDIM + (k - FDIM)];
        sW[idx] = (__bf16)w;
    }
    if (tid < HDIM) sWc[tid] = Wc[tid];
    if (tid == 0) { sWc[HDIM] = Wc[HDIM]; sWc[HDIM + 1] = bc[0]; sBreak = 0; }
    if (tid < 16) sHalted[tid] = 0;

    float bias_g[4];
#pragma unroll
    for (int g = 0; g < 4; ++g) {
        int col = g * HDIM + wg * 16 + ln16;
        bias_g[g] = bih[col] + bhh[col];
    }
    __syncthreads();

    // persistent cell state c: 2 M-tiles (32 batch rows per wave) in VGPRs
    v8f cst[2];
#pragma unroll
    for (int m = 0; m < 2; ++m)
#pragma unroll
        for (int r = 0; r < 8; ++r) cst[m][r] = 0.0f;

    const int gt0  = wv * 2;        // this wave's first global M-tile
    const int wcol = wg * 16;       // h-column base owned by this WG
    int allh = 0;                   // (tid 0 only) all local batches halted?

    for (int t = 0; t < T_STEPS; ++t) {
        const __bf16* hprev = hbuf + (size_t)((t & 1) ^ 1) * (BATCH * HDIM);
        __bf16*       hcur  = hbuf + (size_t)(t & 1) * (BATCH * HDIM);
        const float*  xt    = X + (size_t)t * BATCH * FDIM;

        v8f acc[2][4];
#pragma unroll
        for (int m = 0; m < 2; ++m)
#pragma unroll
            for (int n = 0; n < 4; ++n)
#pragma unroll
                for (int r = 0; r < 8; ++r) acc[m][n][r] = 0.0f;

        // ---- gates[all 256 rows, 64 cols] = [x_t | h_prev] x Wslice --------
#pragma unroll
        for (int kc = 0; kc < 12; ++kc) {
            V16 a[2];
            if (kc < 4) {   // x part (f32 -> bf16 in-register)
                int k0 = kc * 32 + half * 8;
#pragma unroll
                for (int m = 0; m < 2; ++m) {
                    const float* ap = xt + (size_t)((gt0 + m) * 16 + ln16) * FDIM + k0;
                    v4f f0 = *(const v4f*)(ap);
                    v4f f1 = *(const v4f*)(ap + 4);
                    v4f f2 = *(const v4f*)(ap + 16);
                    v4f f3 = *(const v4f*)(ap + 20);
#pragma unroll
                    for (int e = 0; e < 4; ++e) {
                        a[m].e[e]      = (__bf16)f0[e];
                        a[m].e[4 + e]  = (__bf16)f1[e];
                        a[m].e[8 + e]  = (__bf16)f2[e];
                        a[m].e[12 + e] = (__bf16)f3[e];
                    }
                }
            } else {        // h part (bf16 direct)
                int k0 = kc * 32 - FDIM + half * 8;
#pragma unroll
                for (int m = 0; m < 2; ++m) {
                    const __bf16* ap = hprev + (size_t)((gt0 + m) * 16 + ln16) * HDIM + k0;
                    a[m].h[0] = *(const v8bf*)(ap);
                    a[m].h[1] = *(const v8bf*)(ap + 16);
                }
            }
#pragma unroll
            for (int n = 0; n < 4; ++n) {
                V16 bu;
                bu.h[0] = *(const v8bf*)&sW[(n * 16 + ln16) * KDIM + kc * 32 + half * 8];
                bu.h[1] = *(const v8bf*)&sW[(n * 16 + ln16) * KDIM + kc * 32 + 16 + half * 8];
                v16bf b = bu.v;
                acc[0][n] = __builtin_amdgcn_wmma_f32_16x16x32_bf16(
                    false, a[0].v, false, b, (short)0, acc[0][n], false, false);
                acc[1][n] = __builtin_amdgcn_wmma_f32_16x16x32_bf16(
                    false, a[1].v, false, b, (short)0, acc[1][n], false, false);
            }
        }

        // ---- LSTM elementwise; write h slice (bf16) to current buffer -----
#pragma unroll
        for (int m = 0; m < 2; ++m) {
            int rowbase = (gt0 + m) * 16 + half * 8;
#pragma unroll
            for (int r = 0; r < 8; ++r) {
                float ig = acc[m][0][r] + bias_g[0];
                float fg = acc[m][1][r] + bias_g[1];
                float gg = acc[m][2][r] + bias_g[2];
                float og = acc[m][3][r] + bias_g[3];
                float cs = sigf(fg) * cst[m][r] + sigf(ig) * tanh_fast(gg);
                cst[m][r] = cs;
                float hv = sigf(og) * tanh_fast(cs);
                hcur[(size_t)(rowbase + r) * HDIM + wcol + ln16] = (__bf16)hv;
            }
        }

        // ---- device-wide barrier (per-step counter; high bits = halt vote) -
        __threadfence();
        __syncthreads();
        if (tid == 0) {
            unsigned add = 1u + (allh ? 0x10000u : 0u);
            __hip_atomic_fetch_add(&barr[t], add, __ATOMIC_ACQ_REL,
                                   __HIP_MEMORY_SCOPE_AGENT);
            unsigned v;
            do {
                v = __hip_atomic_load(&barr[t], __ATOMIC_ACQUIRE,
                                      __HIP_MEMORY_SCOPE_AGENT);
                if ((v & 0xFFFFu) >= NWG) break;
                __builtin_amdgcn_s_sleep(1);
            } while (true);
            sBreak = ((v >> 16) == NWG) ? 1 : 0;
        }
        __syncthreads();
        if (sBreak) break;   // every batch row already halted & copied

        // ---- controller + halting for this WG's 16 batch rows --------------
        {
            const int bl = tid >> 4, seg = tid & 15;
            const int b  = wg * 16 + bl;
            const __bf16* hr = hcur + (size_t)b * HDIM + seg * 16;
            float part = 0.0f;
#pragma unroll
            for (int j = 0; j < 16; ++j) part += (float)hr[j] * sWc[seg * 16 + j];
            sRed[bl][seg] = part;
            __syncthreads();
            if (seg == 0) {
                float dot = 0.0f;
#pragma unroll
                for (int j = 0; j < 16; ++j) dot += sRed[bl][j];
                float p = sigf(dot + (float)t * sWc[HDIM] + sWc[HDIM + 1]);
                p = (1.0f - EPS_C) * p + EPS_C * 0.05f;
                float uv = U[(size_t)t * BATCH + b];
                int was = sHalted[bl];
                int a   = (uv < p) ? 1 : 0;
                sCopy[bl]   = (!was) && (a || (t == T_STEPS - 1));
                sHalted[bl] = was | a;
            }
            __syncthreads();
            if (sCopy[bl]) {
                float* sp = Shalt + (size_t)b * HDIM + seg * 16;
#pragma unroll
                for (int j = 0; j < 16; ++j) sp[j] = (float)hr[j];
            }
            __syncthreads();
            if (tid == 0) {
                int a2 = 1;
#pragma unroll
                for (int i = 0; i < 16; ++i) a2 &= sHalted[i];
                allh = a2;
            }
        }
    }

    // ---- final classifier: y_hat rows for this WG's 16 batches -------------
    __threadfence();
    __syncthreads();
    if (tid < 16 * NCLS) {
        int b  = wg * 16 + tid / NCLS;
        int cl = tid % NCLS;
        const float* sp = Shalt + (size_t)b * HDIM;
        const float* wr = Wd + (size_t)cl * HDIM;
        float acc = bd[cl];
        for (int j = 0; j < HDIM; ++j) acc += sp[j] * wr[j];
        out[b * NCLS + cl] = acc;
    }
}

// ---------------------------------------------------------------------------
extern "C" void kernel_launch(void* const* d_in, const int* in_sizes, int n_in,
                              void* d_out, int out_size, void* d_ws, size_t ws_size,
                              hipStream_t stream) {
    (void)in_sizes; (void)n_in; (void)out_size; (void)ws_size;
    const float* X   = (const float*)d_in[0];
    const float* U   = (const float*)d_in[1];
    const float* Wih = (const float*)d_in[2];
    const float* Whh = (const float*)d_in[3];
    const float* bih = (const float*)d_in[4];
    const float* bhh = (const float*)d_in[5];
    const float* Wc  = (const float*)d_in[6];
    const float* bc  = (const float*)d_in[7];
    // d_in[8] (W_b), d_in[9] (b_b): baseline net — dead w.r.t. the output
    const float* Wd  = (const float*)d_in[10];
    const float* bd  = (const float*)d_in[11];
    float* out = (float*)d_out;

    char* ws = (char*)d_ws;
    unsigned*       barr  = (unsigned*)ws;                          // 4000 B
    unsigned short* hbuf  = (unsigned short*)(ws + 4096);           // 256 KB bf16 x2
    float*          Shalt = (float*)(ws + 4096 + 2 * BATCH * HDIM * 2); // 256 KB

    earliest_init<<<512, 256, 0, stream>>>(barr, hbuf);
    earliest_main<<<NWG, 256, 0, stream>>>(X, U, Wih, Whh, bih, bhh,
                                           Wc, bc, Wd, bd, out,
                                           hbuf, Shalt, barr);
}
